// DeLaN_62431644614813
// MI455X (gfx1250) — matrix operand order repaired and checked by reading, not compiled, not run on hardware
//
#include <hip/hip_runtime.h>
#include <hip/hip_bf16.h>

#define INP   7
#define HID   512
#define NLAY  4
#define LOWN  21
#define BATCH 8192
#define SLOPE 0.01f
#define BIASC 0.1f
#define SPW   8            // samples per workgroup
#define NCOL  (SPW * 8)    // 64 LDS columns: value + 7 tangents per sample
#define XSTR  520          // padded column stride (halfs): 1040B -> conflict-free
#define HSTR  68           // padded head column stride (floats): 272B

typedef __attribute__((ext_vector_type(16))) _Float16 v16h;
typedef __attribute__((ext_vector_type(8)))  _Float16 h8;
typedef __attribute__((ext_vector_type(8)))  float    v8f;
typedef __attribute__((ext_vector_type(4)))  float    f4;

// ---------------------------------------------------------------------------
// Kernel 0: f32 -> f16 weight conversion; build padded 64x512 head matrix
// rows 0..20 = lo_w, 21..27 = ld_w, 28..34 = g_w, 35..63 = 0
// ---------------------------------------------------------------------------
__global__ void delan_cvt_weights(const float* __restrict__ fc_w,
                                  const float* __restrict__ lo_w,
                                  const float* __restrict__ ld_w,
                                  const float* __restrict__ g_w,
                                  _Float16* __restrict__ fc16,
                                  _Float16* __restrict__ wcat) {
  const int stride = gridDim.x * blockDim.x;
  for (int i = blockIdx.x * blockDim.x + threadIdx.x; i < NLAY * HID * HID; i += stride)
    fc16[i] = (_Float16)fc_w[i];
  for (int j = blockIdx.x * blockDim.x + threadIdx.x; j < 64 * HID; j += stride) {
    int r = j >> 9, c = j & (HID - 1);
    float v = 0.f;
    if (r < LOWN)                 v = lo_w[r * HID + c];
    else if (r < LOWN + INP)      v = ld_w[(r - LOWN) * HID + c];
    else if (r < LOWN + 2 * INP)  v = g_w[(r - LOWN - INP) * HID + c];
    wcat[j] = (_Float16)v;
  }
}

// ---------------------------------------------------------------------------
// Main kernel: fused value + 7-tangent forward (jacfwd) on WMMA, then
// per-sample 7x7 Lagrangian dynamics assembly.
// ---------------------------------------------------------------------------
__global__ __launch_bounds__(256) void delan_main(
    const float* __restrict__ state,
    const float* __restrict__ pre_w, const float* __restrict__ pre_b,
    const _Float16* __restrict__ fc16, const float* __restrict__ fc_b,
    const _Float16* __restrict__ wcat,
    const float* __restrict__ lo_b, const float* __restrict__ ld_b,
    const float* __restrict__ g_b,
    float* __restrict__ out)
{
  __shared__ __align__(16) _Float16 xbuf[2][NCOL * XSTR]; // 2 x 65 KB ping-pong
  __shared__ __align__(16) float    hbuf[NCOL * HSTR];    // 17 KB head output

  const int  tid    = threadIdx.x;
  const int  lane   = tid & 31;
  const int  wv     = tid >> 5;
  const int  laneM  = lane & 15;
  const bool hiHalf = lane >= 16;
  const int  s0     = blockIdx.x * SPW;
  const v8f  vzero  = {0.f, 0.f, 0.f, 0.f, 0.f, 0.f, 0.f, 0.f};

  // ---- Phase 0: input layer 7->512 (tiny, plain VALU) + tangent seed ----
  for (int idx = tid; idx < SPW * HID; idx += 256) {
    const int s = idx >> 9;
    const int h = idx & (HID - 1);
    const float* st = state + (size_t)(s0 + s) * (3 * INP);
    float w[INP];
    float z = pre_b[h];
#pragma unroll
    for (int j = 0; j < INP; ++j) { w[j] = pre_w[h * INP + j]; z = fmaf(w[j], st[j], z); }
    const float f = z > 0.f ? 1.f : SLOPE;
    _Float16* xc = &xbuf[0][(8 * s) * XSTR + h];
    xc[0] = (_Float16)(z * f);                               // leaky_relu(z)
#pragma unroll
    for (int t = 1; t < 8; ++t) xc[t * XSTR] = (_Float16)(f * w[t - 1]); // dX/dq
  }
  __syncthreads();

  // ---- Phase 1: four 512x512 layers as WMMA GEMMs over 64 columns ----
  int cur = 0;
  for (int l = 0; l < NLAY; ++l) {
    const _Float16* __restrict__ W    = fc16 + (size_t)l * HID * HID;
    const float*    __restrict__ bias = fc_b + l * HID;
    const _Float16* Xc = xbuf[cur];
    _Float16*       Yc = xbuf[cur ^ 1];

    for (int mt = 0; mt < 4; ++mt) {
      const int rowBase = wv * 64 + mt * 16;
      v8f acc[4];
#pragma unroll
      for (int nt = 0; nt < 4; ++nt) acc[nt] = vzero;

      const _Float16* Wr = W + (size_t)(rowBase + laneM) * HID;
      for (int ks = 0; ks < 16; ++ks) {
        const int k0 = ks * 32;
        // A fragment: 16-bit 16x32 layout (lanes<16: K 0-7 & 16-23; lanes>=16: +8)
        const int ka = k0 + (hiHalf ? 8 : 0);
        const h8 alo = *(const h8*)(Wr + ka);
        const h8 ahi = *(const h8*)(Wr + ka + 16);
        v16h a;
#pragma unroll
        for (int i = 0; i < 8; ++i) { a[i] = alo[i]; a[i + 8] = ahi[i]; }
        // B fragment: same K layout as A (lanes<16: K 0-7 & 16-23; lanes>=16: +8)
        const int kb = k0 + (hiHalf ? 8 : 0);
#pragma unroll
        for (int nt = 0; nt < 4; ++nt) {
          const _Float16* xp = Xc + (nt * 16 + laneM) * XSTR + kb;
          const h8 blo = *(const h8*)xp;
          const h8 bhi = *(const h8*)(xp + 16);
          v16h b;
#pragma unroll
          for (int i = 0; i < 8; ++i) { b[i] = blo[i]; b[i + 8] = bhi[i]; }
          acc[nt] = __builtin_amdgcn_wmma_f32_16x16x32_f16(
              false, a, false, b, (short)0, acc[nt], false, false);
        }
      }

      // Epilogue: col t=0 -> leaky_relu(z+b); cols t>=1 -> lrelu'(z0+b) * z
      const int rowOff = rowBase + (hiHalf ? 8 : 0);
      const float* bp  = bias + rowOff;
      const int srcL   = lane & 24; // t=0 lane of this 8-column sample group
#pragma unroll
      for (int nt = 0; nt < 4; ++nt) {
        h8 rv;
#pragma unroll
        for (int v = 0; v < 8; ++v) {
          const float z0  = __shfl(acc[nt][v], srcL, 32);
          const float zb  = z0 + bp[v];
          const float f   = zb > 0.f ? 1.f : SLOPE;
          const float val = ((lane & 7) == 0) ? zb * f : acc[nt][v] * f;
          rv[v] = (_Float16)val;
        }
        *(h8*)(Yc + (nt * 16 + laneM) * XSTR + rowOff) = rv;
      }
    }
    __syncthreads();
    cur ^= 1;
  } // final activations+tangents live in xbuf[0]

  // ---- Phase 2: head GEMM 64x512 @ 512x64 -> raw f32 pre-activations ----
  {
    const _Float16* Xc = xbuf[0];
    const int mt      = wv >> 1;
    const int ntBase  = (wv & 1) * 2;
    const int rowBase = mt * 16;
    v8f acc[2] = {vzero, vzero};
    const _Float16* Wr = wcat + (size_t)(rowBase + laneM) * HID;
    for (int ks = 0; ks < 16; ++ks) {
      const int k0 = ks * 32;
      const int ka = k0 + (hiHalf ? 8 : 0);
      const h8 alo = *(const h8*)(Wr + ka);
      const h8 ahi = *(const h8*)(Wr + ka + 16);
      v16h a;
#pragma unroll
      for (int i = 0; i < 8; ++i) { a[i] = alo[i]; a[i + 8] = ahi[i]; }
      const int kb = k0 + (hiHalf ? 8 : 0);
#pragma unroll
      for (int nn = 0; nn < 2; ++nn) {
        const _Float16* xp = Xc + ((ntBase + nn) * 16 + laneM) * XSTR + kb;
        const h8 blo = *(const h8*)xp;
        const h8 bhi = *(const h8*)(xp + 16);
        v16h b;
#pragma unroll
        for (int i = 0; i < 8; ++i) { b[i] = blo[i]; b[i + 8] = bhi[i]; }
        acc[nn] = __builtin_amdgcn_wmma_f32_16x16x32_f16(
            false, a, false, b, (short)0, acc[nn], false, false);
      }
    }
    const int rowOff = rowBase + (hiHalf ? 8 : 0);
#pragma unroll
    for (int nn = 0; nn < 2; ++nn) {
      float* hp = hbuf + ((ntBase + nn) * 16 + laneM) * HSTR + rowOff;
      f4 lo4 = {acc[nn][0], acc[nn][1], acc[nn][2], acc[nn][3]};
      f4 hi4 = {acc[nn][4], acc[nn][5], acc[nn][6], acc[nn][7]};
      *(f4*)hp       = lo4;
      *(f4*)(hp + 4) = hi4;
    }
  }
  __syncthreads();

  // ---- Phase 3: per-sample 7x7 dynamics assembly (1 thread / sample) ----
  if (tid < SPW) {
    const int s  = tid;
    const int gs = s0 + s;
    const float* st = state + (size_t)gs * (3 * INP);
    float qd[INP], qdd[INP];
#pragma unroll
    for (int k = 0; k < INP; ++k) { qd[k] = st[INP + k]; qdd[k] = st[2 * INP + k]; }
    const float* P = hbuf + (8 * s) * HSTR;     // P[t*HSTR + r] = raw head output
    float* Lq = (float*)xbuf[1] + s * 512;      // reuse free LDS: L_q[i][j][k]
    for (int i = 0; i < 343; ++i) Lq[i] = 0.f;
    float Lm[49];
    for (int i = 0; i < 49; ++i) Lm[i] = 0.f;

    // strict lower triangle (tril_indices order), linear head -> Jacobian raw
    int m = 0;
    for (int i = 1; i < INP; ++i)
      for (int j = 0; j < i; ++j) {
        Lm[i * 7 + j] = P[m] + lo_b[m];
        for (int k = 0; k < INP; ++k) Lq[(i * 7 + j) * 7 + k] = P[(1 + k) * HSTR + m];
        ++m;
      }
    // diagonal: softplus(z)+BIAS; d/dq = sigmoid(z) * (ld_w @ dX)
    float gv[INP];
    for (int k = 0; k < INP; ++k) {
      const float z  = P[LOWN + k] + ld_b[k];
      const float sp = fmaxf(z, 0.f) + log1pf(expf(-fabsf(z)));
      Lm[k * 7 + k]  = sp + BIASC;
      const float sg = 1.f / (1.f + expf(-z));
      for (int j = 0; j < INP; ++j) Lq[(k * 7 + k) * 7 + j] = sg * P[(1 + j) * HSTR + LOWN + k];
      gv[k] = P[LOWN + INP + k] + g_b[k];
    }

    // Ldot[i][j] = sum_k Lq[i][j][k] * qd[k]
    float Ldot[49];
    for (int i = 0; i < INP; ++i)
      for (int j = 0; j < INP; ++j) {
        float a2 = 0.f;
        for (int k = 0; k < INP; ++k) a2 = fmaf(Lq[(i * 7 + j) * 7 + k], qd[k], a2);
        Ldot[i * 7 + j] = a2;
      }
    // av = L^T qd ; wv2 = Ldot^T qd
    float av[INP], wv2[INP];
    for (int m2 = 0; m2 < INP; ++m2) {
      float aa = 0.f, ww = 0.f;
      for (int j = 0; j < INP; ++j) {
        aa = fmaf(qd[j], Lm[j * 7 + m2], aa);
        ww = fmaf(qd[j], Ldot[j * 7 + m2], ww);
      }
      av[m2] = aa; wv2[m2] = ww;
    }
    // c = c1 - 0.5*c2 ;  c2[k] = 2 * sum_m av[m] * (sum_i qd[i]*Lq[i][m][k])
    float cc[INP];
    for (int k = 0; k < INP; ++k) cc[k] = 0.f;
    for (int m2 = 0; m2 < INP; ++m2)
      for (int k = 0; k < INP; ++k) {
        float bq = 0.f;
        for (int i = 0; i < INP; ++i) bq = fmaf(qd[i], Lq[(i * 7 + m2) * 7 + k], bq);
        cc[k] -= av[m2] * bq;       // accumulates -0.5*c2
      }
    for (int i = 0; i < INP; ++i) { // + c1 = L*wv2 + Ldot*av
      float c1 = 0.f;
      for (int m2 = 0; m2 < INP; ++m2)
        c1 = fmaf(Lm[i * 7 + m2], wv2[m2], fmaf(Ldot[i * 7 + m2], av[m2], c1));
      cc[i] += c1;
    }
    // H = L L^T ;  u = L (L^T qdd) + c + g
    float Hm[49];
    for (int i = 0; i < INP; ++i)
      for (int j = 0; j < INP; ++j) {
        float hh = 0.f;
        for (int m2 = 0; m2 < INP; ++m2) hh = fmaf(Lm[i * 7 + m2], Lm[j * 7 + m2], hh);
        Hm[i * 7 + j] = hh;
      }
    float ev[INP];
    for (int m2 = 0; m2 < INP; ++m2) {
      float e = 0.f;
      for (int j = 0; j < INP; ++j) e = fmaf(qdd[j], Lm[j * 7 + m2], e);
      ev[m2] = e;
    }

    float* out_u = out;
    float* out_H = out + (size_t)BATCH * INP;
    float* out_c = out + (size_t)BATCH * (INP + INP * INP);
    float* out_g = out + (size_t)BATCH * (2 * INP + INP * INP);
    for (int i = 0; i < INP; ++i) {
      float u = 0.f;
      for (int m2 = 0; m2 < INP; ++m2) u = fmaf(Lm[i * 7 + m2], ev[m2], u);
      out_u[(size_t)gs * INP + i] = u + cc[i] + gv[i];
      out_c[(size_t)gs * INP + i] = cc[i];
      out_g[(size_t)gs * INP + i] = gv[i];
    }
    for (int i = 0; i < 49; ++i) out_H[(size_t)gs * 49 + i] = Hm[i];
  }
}

// ---------------------------------------------------------------------------
extern "C" void kernel_launch(void* const* d_in, const int* in_sizes, int n_in,
                              void* d_out, int out_size, void* d_ws, size_t ws_size,
                              hipStream_t stream) {
  (void)in_sizes; (void)n_in; (void)out_size; (void)ws_size;
  const float* state = (const float*)d_in[0];
  const float* pre_w = (const float*)d_in[1];
  const float* pre_b = (const float*)d_in[2];
  const float* fc_w  = (const float*)d_in[3];
  const float* fc_b  = (const float*)d_in[4];
  const float* lo_w  = (const float*)d_in[5];
  const float* lo_b  = (const float*)d_in[6];
  const float* ld_w  = (const float*)d_in[7];
  const float* ld_b  = (const float*)d_in[8];
  const float* g_w   = (const float*)d_in[9];
  const float* g_b   = (const float*)d_in[10];

  _Float16* fc16 = (_Float16*)d_ws;                       // 4*512*512 halfs = 2 MB
  _Float16* wcat = fc16 + (size_t)NLAY * HID * HID;       // 64*512 halfs

  delan_cvt_weights<<<256, 256, 0, stream>>>(fc_w, lo_w, ld_w, g_w, fc16, wcat);
  delan_main<<<BATCH / SPW, 256, 0, stream>>>(state, pre_w, pre_b, fc16, fc_b,
                                              wcat, lo_b, ld_b, g_b, (float*)d_out);
}
